// EncoderRNN_23862838296990
// MI455X (gfx1250) — matrix operand order repaired and checked by reading, not compiled, not run on hardware
//
#include <hip/hip_runtime.h>

// ---------------------------------------------------------------------------
// Bidirectional LSTM encoder for MI455X (gfx1250, wave32, WMMA, async-to-LDS).
//   B=64, S=512, V=32000, E=512, H=512.
// Prep:    emb -> bf16 copy; Wf/Wb/Uf/Ub -> bf16 transposed [N=2048][K=512].
// Phase 1: input GEMM (x@W+b) with v_wmma_f32_16x16x32_bf16, double-buffered
//          global_load_async_to_lds_b128 staging; Z stored bf16 (2x128 MB).
// Phase 2: persistent 32-WG recurrence; U slice + h broadcast staged to LDS
//          with async b128 copies; grid-wide atomic barrier per timestep.
// ---------------------------------------------------------------------------

#define B_   64
#define S_   512
#define E_   512
#define H_   512
#define G4_  2048   // 4*H
#define NWG  32

typedef __attribute__((ext_vector_type(16))) __bf16 v16bf;
typedef __attribute__((ext_vector_type(8)))  __bf16 v8bf;
typedef __attribute__((ext_vector_type(4)))  __bf16 v4bf;
typedef __attribute__((ext_vector_type(8)))  float  v8f;

// --- CDNA5 async global->LDS copy (ASYNCcnt) --------------------------------
// Flat LDS pointer: aperture in [63:32], allocation-relative offset in [31:0],
// so truncation yields the LDS byte offset GLOBAL_LOAD_ASYNC_TO_LDS expects.
__device__ __forceinline__ void async_b128(void* lds_dst, const void* gsrc) {
  unsigned loff = (unsigned)(uintptr_t)lds_dst;
  asm volatile("global_load_async_to_lds_b128 %0, %1, off"
               :: "v"(loff), "v"(gsrc) : "memory");
}
__device__ __forceinline__ void async_b128_off(unsigned loff, const void* gsrc) {
  asm volatile("global_load_async_to_lds_b128 %0, %1, off"
               :: "v"(loff), "v"(gsrc) : "memory");
}
__device__ __forceinline__ void wait_async0() {
  asm volatile("s_wait_asynccnt 0x0" ::: "memory");
}
__device__ __forceinline__ void wait_async3() {
  asm volatile("s_wait_asynccnt 0x3" ::: "memory");
}

// Build a v16bf A/B fragment from two 8-element (16B-aligned) LDS chunks.
__device__ __forceinline__ v16bf ld_frag(const __bf16* p0, const __bf16* p1) {
  v8bf lo = *(const v8bf*)p0;
  v8bf hi = *(const v8bf*)p1;
  v16bf r;
#pragma unroll
  for (int i = 0; i < 8; ++i) { r[i] = lo[i]; r[i + 8] = hi[i]; }
  return r;
}

__device__ __forceinline__ float sigmoidf_(float x) {
  return 1.0f / (1.0f + __expf(-x));
}

// ---------------------------------------------------------------------------
// Kernel 0: zero h buffers + barrier words.
// ---------------------------------------------------------------------------
__global__ void init_state(unsigned* __restrict__ p, int n) {
  int i = blockIdx.x * blockDim.x + threadIdx.x;
  if (i < n) p[i] = 0u;
}

// ---------------------------------------------------------------------------
// Kernel P1: emb fp32 -> bf16 (same layout), vectorized x4.
// ---------------------------------------------------------------------------
__global__ void __launch_bounds__(256)
cvt_emb(const float* __restrict__ src, __bf16* __restrict__ dst, long n4) {
  long i = (long)blockIdx.x * 256 + threadIdx.x;
  if (i < n4) {
    float4 v = ((const float4*)src)[i];
    v4bf o;
    o[0] = (__bf16)v.x; o[1] = (__bf16)v.y;
    o[2] = (__bf16)v.z; o[3] = (__bf16)v.w;
    *(v4bf*)(dst + i * 4) = o;
  }
}

// ---------------------------------------------------------------------------
// Kernel P2: fp32 [512][2048] -> bf16 transposed [2048][512], LDS-tiled 64x64.
// grid = (8, 32, 4): z selects {Wf, Wb, Uf, Ub}.
// ---------------------------------------------------------------------------
__global__ void __launch_bounds__(256)
transpose_cvt(const float* __restrict__ Wf, const float* __restrict__ Wb,
              const float* __restrict__ Uf, const float* __restrict__ Ub,
              __bf16* __restrict__ WfT, __bf16* __restrict__ WbT,
              __bf16* __restrict__ UfT, __bf16* __restrict__ UbT) {
  __shared__ __bf16 tile[64 * 72];   // pad 8 to dodge bank conflicts
  const float* src; __bf16* dst;
  switch (blockIdx.z) {
    case 0:  src = Wf; dst = WfT; break;
    case 1:  src = Wb; dst = WbT; break;
    case 2:  src = Uf; dst = UfT; break;
    default: src = Ub; dst = UbT; break;
  }
  const int k0 = blockIdx.x * 64, n0 = blockIdx.y * 64;
  for (int idx = threadIdx.x; idx < 4096; idx += 256) {
    int kk = idx >> 6, nn = idx & 63;                         // coalesced read
    tile[nn * 72 + kk] = (__bf16)src[(long)(k0 + kk) * G4_ + n0 + nn];
  }
  __syncthreads();
  for (int idx = threadIdx.x; idx < 4096; idx += 256) {
    int nn = idx >> 6, kk = idx & 63;                         // coalesced write
    dst[(long)(n0 + nn) * E_ + k0 + kk] = tile[nn * 72 + kk];
  }
}

// ---------------------------------------------------------------------------
// Kernel 1: Z[dir][s][b][n] = bf16( embbf[tok[b,s]] @ W[dir]^T + bias[dir] )
// All-bf16 staging via async b128, double-buffered. All global source
// pointers are hoisted out of the K-stage loop and bumped by 64 B/stage, so
// the steady state is 3 async instr/wave + wait + barrier + 10 ds_load_b128
// + 4 WMMA per stage with no LDS round-trips for addressing.
// ---------------------------------------------------------------------------
__global__ void __launch_bounds__(256)
embed_input_gemm(const int* __restrict__ tokens,
                 const __bf16* __restrict__ embbf,
                 const __bf16* __restrict__ WfT, const float* __restrict__ bf,
                 const __bf16* __restrict__ WbT, const float* __restrict__ bb,
                 __bf16* __restrict__ Zf, __bf16* __restrict__ Zb) {
  __shared__ __bf16 At[2][64 * 32];    // A tile [row][k]
  __shared__ __bf16 Bt[2][128 * 32];   // B tile [n][k] (WT is already [n][k])
  __shared__ int    tok[64];

  const int tid  = threadIdx.x;
  const int lane = tid & 31;
  const int wave = tid >> 5;

  int bx = blockIdx.x;
  const int dir = bx / (S_ * 16);
  bx -= dir * (S_ * 16);
  const int s  = bx >> 4;            // M-tile == one timestep (rows = batch)
  const int n0 = (bx & 15) * 128;    // N-tile base in [0,2048)

  const __bf16* WT  = dir ? WbT : WfT;
  const float*  bia = dir ? bb : bf;
  __bf16*       Z   = dir ? Zb : Zf;

  if (tid < 64) tok[tid] = tokens[tid * S_ + s];
  __syncthreads();

  // Per-thread staging assignment (fixed across stages):
  //   At chunk: row r = tid/4, quarter q = tid%4      (one 16B chunk)
  //   Bt chunks: rows tid/4 and 64+tid/4, quarter tid%4 (two 16B chunks)
  const int rA = tid >> 2, qA = tid & 3;
  const int cB = tid >> 1, qB = tid & 1;
  const __bf16* aSrc = embbf + (long)tok[rA] * E_ + qA * 8;
  const __bf16* bSrc0 = WT + (long)(n0 + cB) * E_ + qB * 8;
  const __bf16* bSrc1 = bSrc0 + (long)128 * E_;
  const unsigned ldsA0 = (unsigned)(uintptr_t)&At[0][rA * 32 + qA * 8];
  const unsigned ldsA1 = (unsigned)(uintptr_t)&At[1][rA * 32 + qA * 8];
  const unsigned ldsB0 = (unsigned)(uintptr_t)&Bt[0][cB * 32 + qB * 8];
  const unsigned ldsB1 = (unsigned)(uintptr_t)&Bt[1][cB * 32 + qB * 8];
  const unsigned bHi   = (128 - cB + (tid >> 1)) * 0 + 128 * 32 * (unsigned)sizeof(__bf16) / 1; // unused guard
  (void)bHi;
  const unsigned ldsB0b = ldsB0 + 128 * 32 * (unsigned)sizeof(__bf16);  // row 128+cB... 
  const unsigned ldsB1b = ldsB1 + 128 * 32 * (unsigned)sizeof(__bf16);

  // NOTE: Bt rows are [c][k] with c in [0,128); chunk for row 128+cB does not
  // exist -- second B chunk covers rows [128,256)?? No: Bt has 128 rows; the
  // two chunks per thread are rows cB (i=0) and cB+... re-derive:
  //   512 chunks total = 128 rows x 2 halves x ... = 256 (row,half) pairs?
  // 128 rows * 32 bf16 = 4096 elems = 512 x 8-elem chunks; per row 4 chunks.
  // Assign: chunk idx = tid (0..255) and tid+256: row = idx>>2, q = idx&3.
  // (Fixed below; the ldsB*/bSrc* values above are recomputed accordingly.)

  const int cB0 = tid >> 2, qB0 = tid & 3;          // chunk tid
  const int cB1 = (tid + 256) >> 2, qB1 = tid & 3;  // chunk tid+256
  const __bf16* bS0 = WT + (long)(n0 + cB0) * E_ + qB0 * 8;
  const __bf16* bS1 = WT + (long)(n0 + cB1) * E_ + qB1 * 8;
  const unsigned lB0[2] = {(unsigned)(uintptr_t)&Bt[0][cB0 * 32 + qB0 * 8],
                           (unsigned)(uintptr_t)&Bt[1][cB0 * 32 + qB0 * 8]};
  const unsigned lB1[2] = {(unsigned)(uintptr_t)&Bt[0][cB1 * 32 + qB1 * 8],
                           (unsigned)(uintptr_t)&Bt[1][cB1 * 32 + qB1 * 8]};
  const unsigned lA[2]  = {ldsA0, ldsA1};

  v8f acc[4] = {};
  const int nn    = (wave << 4) + (lane & 15);
  const int koffB = (lane < 16) ? 0 : 16;
  const int kbA   = (lane < 16) ? 0 : 16;

  // Prologue: stage 0 into buffer 0. Each stage advances K by 32 elems (64 B).
  async_b128_off(lA[0], aSrc);
  async_b128_off(lB0[0], bS0);
  async_b128_off(lB1[0], bS1);
  aSrc += 32; bS0 += 32; bS1 += 32;

  for (int st = 0; st < 16; ++st) {
    const int buf = st & 1;
    if (st + 1 < 16) {
      const int nb = buf ^ 1;
      async_b128_off(lA[nb], aSrc);
      async_b128_off(lB0[nb], bS0);
      async_b128_off(lB1[nb], bS1);
      aSrc += 32; bS0 += 32; bS1 += 32;
      wait_async3();
    } else {
      wait_async0();
    }
    __syncthreads();

    v16bf bfrag = ld_frag(&Bt[buf][nn * 32 + koffB],
                          &Bt[buf][nn * 32 + koffB + 8]);
#pragma unroll
    for (int mt = 0; mt < 4; ++mt) {
      const int row = (mt << 4) + (lane & 15);
      v16bf afrag = ld_frag(&At[buf][row * 32 + kbA],
                            &At[buf][row * 32 + kbA + 8]);
      acc[mt] = __builtin_amdgcn_wmma_f32_16x16x32_bf16(
          false, afrag, false, bfrag, (short)0, acc[mt], false, false);
    }
    __syncthreads();
  }

  // D layout: VGPR r -> (M=r, N=lane) lanes 0-15 ; (M=r+8, N=lane-16) 16-31
  const int nc = n0 + (wave << 4) + (lane & 15);
  const float bv = bia[nc];
#pragma unroll
  for (int mt = 0; mt < 4; ++mt) {
#pragma unroll
    for (int r = 0; r < 8; ++r) {
      int brow = (mt << 4) + r + ((lane >> 4) << 3);   // batch index
      Z[((long)s * B_ + brow) * G4_ + nc] = (__bf16)(acc[mt][r] + bv);
    }
  }
}

// ---------------------------------------------------------------------------
// Grid-wide barrier across the 32 persistent WGs (agent-scope acq/rel).
// ---------------------------------------------------------------------------
__device__ __forceinline__ void gbarrier(unsigned* cnt, unsigned* gen) {
  __syncthreads();
  __builtin_amdgcn_fence(__ATOMIC_RELEASE, "agent");
  if (threadIdx.x == 0) {
    unsigned g = __hip_atomic_load(gen, __ATOMIC_ACQUIRE, __HIP_MEMORY_SCOPE_AGENT);
    unsigned prev =
        __hip_atomic_fetch_add(cnt, 1u, __ATOMIC_ACQ_REL, __HIP_MEMORY_SCOPE_AGENT);
    if (prev == NWG - 1u) {
      __hip_atomic_store(cnt, 0u, __ATOMIC_RELAXED, __HIP_MEMORY_SCOPE_AGENT);
      __hip_atomic_fetch_add(gen, 1u, __ATOMIC_RELEASE, __HIP_MEMORY_SCOPE_AGENT);
    } else {
      while (__hip_atomic_load(gen, __ATOMIC_ACQUIRE, __HIP_MEMORY_SCOPE_AGENT) == g)
        __builtin_amdgcn_s_sleep(2);
    }
  }
  __syncthreads();
  __builtin_amdgcn_fence(__ATOMIC_ACQUIRE, "agent");  // invalidate stale L0 lines
}

// ---------------------------------------------------------------------------
// Kernel 2: persistent bidirectional LSTM recurrence.
// blockIdx 0..15 = forward slices, 16..31 = backward slices.
// Dynamic LDS (232 KB): Ut[128][512] bf16 | hs[64][512] bf16 |
//                       zs[64][128] f32   | cst[64*32] f32
// ---------------------------------------------------------------------------
__global__ void __launch_bounds__(256)
bilstm_recurrent(const __bf16* __restrict__ Zf, const __bf16* __restrict__ Zb,
                 const __bf16* __restrict__ UfT, const __bf16* __restrict__ UbT,
                 __bf16* hf, __bf16* hb, float* __restrict__ out,
                 unsigned* bar_cnt, unsigned* bar_gen) {
  extern __shared__ char smem[];
  __bf16* Ut  = (__bf16*)smem;                               // 131072 B
  __bf16* hs  = (__bf16*)(smem + 131072);                    //  65536 B
  float*  zs  = (float*)(smem + 131072 + 65536);             //  32768 B
  float*  cst = (float*)(smem + 131072 + 65536 + 32768);     //   8192 B

  const int tid  = threadIdx.x;
  const int lane = tid & 31;
  const int wave = tid >> 5;
  const int dir  = blockIdx.x >> 4;
  const int j0   = (blockIdx.x & 15) * 32;

  const __bf16* UT = dir ? UbT : UfT;
  const __bf16* Z  = dir ? Zb : Zf;
  __bf16*       hg = dir ? hb : hf;

  // One-time: async-stage this WG's U slice (already bf16 + transposed).
  // Ut row c (= gate*32 + jl) <- UT row gcol = gate*512 + j0 + jl.
  for (int idx = tid; idx < 8192; idx += 256) {   // 256 rows x 32 chunks(16B)
    int c = idx >> 5, q = idx & 31;
    int gcol = ((c >> 5) << 9) + j0 + (c & 31);
    async_b128(&Ut[c * H_ + q * 8], UT + (long)gcol * E_ + q * 8);
  }
  for (int p = tid; p < B_ * 32; p += 256) cst[p] = 0.f;
  wait_async0();
  __syncthreads();

  const int nn    = (wave << 4) + (lane & 15);
  const int koffB = (lane < 16) ? 0 : 16;
  const int kbA   = (lane < 16) ? 0 : 16;

  for (int step = 0; step < S_; ++step) {
    const int t = dir ? (S_ - 1 - step) : step;

    // Stage h broadcast (64x512 bf16 = 4096 x 16B) via async copies.
    for (int idx = tid; idx < 4096; idx += 256)
      async_b128((char*)hs + (size_t)idx * 16, (const char*)hg + (size_t)idx * 16);
    wait_async0();
    __syncthreads();

    // zs[64,128] = hs[64,512] @ Ut^T   (512 WMMAs per WG per step)
    v8f acc[4] = {};
    for (int k = 0; k < H_; k += 32) {
      v16bf bfrag =
          ld_frag(&Ut[nn * H_ + k + koffB], &Ut[nn * H_ + k + koffB + 8]);
#pragma unroll
      for (int mt = 0; mt < 4; ++mt) {
        const int row = (mt << 4) + (lane & 15);
        v16bf afrag =
            ld_frag(&hs[row * H_ + k + kbA], &hs[row * H_ + k + kbA + 8]);
        acc[mt] = __builtin_amdgcn_wmma_f32_16x16x32_bf16(
            false, afrag, false, bfrag, (short)0, acc[mt], false, false);
      }
    }
#pragma unroll
    for (int mt = 0; mt < 4; ++mt) {
#pragma unroll
      for (int r = 0; r < 8; ++r) {
        int row = (mt << 4) + r + ((lane >> 4) << 3);
        zs[row * 128 + nn] = acc[mt][r];
      }
    }
    __syncthreads();

    // Gates + state update: thread owns (b, jl); i,f,g,o at zs col g*32+jl.
    const __bf16* Zt = Z + (long)t * B_ * G4_;
    const int t2 = dir ? (t - 1) : (t + 1);
    const __bf16* Zt2 = (t2 >= 0 && t2 < S_) ? (Z + (long)t2 * B_ * G4_) : Zt;
    for (int p = tid; p < B_ * 32; p += 256) {
      int b = p >> 5, jl = p & 31;
      const __bf16* zrow = Zt + (long)b * G4_ + j0 + jl;
      float zi = zs[b * 128 +       jl] + (float)zrow[0];
      float zf = zs[b * 128 +  32 + jl] + (float)zrow[512];
      float zg = zs[b * 128 +  64 + jl] + (float)zrow[1024];
      float zo = zs[b * 128 +  96 + jl] + (float)zrow[1536];
      float ig = sigmoidf_(zi);
      float fg = sigmoidf_(zf);
      float gg = tanhf(zg);
      float og = sigmoidf_(zo);
      float c  = fg * cst[p] + ig * gg;
      cst[p]   = c;
      float h  = og * tanhf(c);
      hg[b * H_ + j0 + jl] = (__bf16)h;
      out[((long)b * S_ + t) * (2 * H_) + dir * H_ + j0 + jl] = h;
      __builtin_prefetch((const void*)(Zt2 + (long)b * G4_ + j0 + jl), 0, 1);
    }

    gbarrier(bar_cnt, bar_gen);
  }
}

// ---------------------------------------------------------------------------
extern "C" void kernel_launch(void* const* d_in, const int* in_sizes, int n_in,
                              void* d_out, int out_size, void* d_ws, size_t ws_size,
                              hipStream_t stream) {
  const int*   tokens = (const int*)d_in[0];
  const float* emb    = (const float*)d_in[1];
  const float* Wf     = (const float*)d_in[2];
  const float* Uf     = (const float*)d_in[3];
  const float* bf     = (const float*)d_in[4];
  const float* Wb     = (const float*)d_in[5];
  const float* Ub     = (const float*)d_in[6];
  const float* bb     = (const float*)d_in[7];
  float* out = (float*)d_out;

  char* ws = (char*)d_ws;
  const size_t ZBYTES = (size_t)S_ * B_ * G4_ * sizeof(__bf16);   // 128 MB
  __bf16* Zf    = (__bf16*)ws;
  __bf16* Zb    = (__bf16*)(ws + ZBYTES);
  char*   aux   = ws + 2 * ZBYTES;
  __bf16* embbf = (__bf16*)aux;                              // 32,768,000 B
  char*   tmats = aux + 33554432;                            // (padded 32 MiB)
  __bf16* WfT = (__bf16*)(tmats);                            // 2 MiB each
  __bf16* WbT = (__bf16*)(tmats + 1 * 2097152);
  __bf16* UfT = (__bf16*)(tmats + 2 * 2097152);
  __bf16* UbT = (__bf16*)(tmats + 3 * 2097152);
  char*   st  = tmats + 4 * 2097152;
  __bf16* hf  = (__bf16*)st;                                 // 64 KB
  __bf16* hb  = (__bf16*)(st + 65536);                       // 64 KB
  unsigned* bar_cnt = (unsigned*)(st + 131072);
  unsigned* bar_gen = (unsigned*)(st + 131072 + 128);

  // Zero h state + barrier words.
  {
    int nwords = (131072 + 256) / 4;
    init_state<<<dim3((nwords + 255) / 256), dim3(256), 0, stream>>>(
        (unsigned*)st, nwords);
  }

  // Prep: bf16 embedding copy + transposed bf16 weights.
  {
    long n4 = (long)32000 * E_ / 4;                          // 4,096,000
    cvt_emb<<<dim3((unsigned)((n4 + 255) / 256)), dim3(256), 0, stream>>>(
        emb, embbf, n4);
    transpose_cvt<<<dim3(E_ / 64, G4_ / 64, 4), dim3(256), 0, stream>>>(
        Wf, Wb, Uf, Ub, WfT, WbT, UfT, UbT);
  }

  // Phase 1: both directions' input GEMMs (2 * 512 * 16 tiles).
  embed_input_gemm<<<dim3(2 * S_ * 16), dim3(256), 0, stream>>>(
      tokens, embbf, WfT, bf, WbT, bb, Zf, Zb);

  // Phase 2: persistent recurrence, 32 WGs, 232 KB dynamic LDS each.
  size_t shmem = 131072 + 65536 + 32768 + 8192;
  bilstm_recurrent<<<dim3(NWG), dim3(256), shmem, stream>>>(
      Zf, Zb, UfT, UbT, hf, hb, out, bar_cnt, bar_gen);
}
